// AttentionInfusionBlock_17609365914149
// MI455X (gfx1250) — matrix operand order repaired and checked by reading, not compiled, not run on hardware
//
#include <hip/hip_runtime.h>

// ---------------------------------------------------------------------------
// Types for CDNA5 WMMA
// ---------------------------------------------------------------------------
typedef _Float16 half8 __attribute__((ext_vector_type(8)));
typedef _Float16 v16h  __attribute__((ext_vector_type(16)));
typedef float    v8f   __attribute__((ext_vector_type(8)));

static constexpr float kScale = 0.125f;  // D^-0.5, D=64

// ---------------------------------------------------------------------------
// fp32 -> fp16 conversion (grid stride)
// ---------------------------------------------------------------------------
__global__ void cvt_f16_kernel(const float* __restrict__ s,
                               _Float16* __restrict__ d, long long n) {
  long long i = (long long)blockIdx.x * blockDim.x + threadIdx.x;
  long long st = (long long)gridDim.x * blockDim.x;
  for (; i < n; i += st) d[i] = (_Float16)s[i];
}

// fp32 [K,N] row-major  ->  fp16 [N,K] row-major (K-major weights so every
// WMMA B-fragment becomes two contiguous b128 loads).
__global__ void cvt_t_f16_kernel(const float* __restrict__ s,
                                 _Float16* __restrict__ d, int K, int N) {
  const long long i = (long long)blockIdx.x * blockDim.x + threadIdx.x;
  const long long total = (long long)K * N;
  if (i < total) {
    const int k = (int)(i % K);
    const int n = (int)(i / K);
    d[i] = (_Float16)s[(long long)k * N + n];
  }
}

__global__ void zero_kernel(float* __restrict__ p, long long n) {
  long long i = (long long)blockIdx.x * blockDim.x + threadIdx.x;
  if (i < n) p[i] = 0.0f;
}

__global__ void finalize_kernel(const float* __restrict__ acc,
                                float* __restrict__ out, long long n) {
  long long i = (long long)blockIdx.x * blockDim.x + threadIdx.x;
  if (i < n) out[i] = acc[i] * (1.0f / 3.0f);
}

// ---------------------------------------------------------------------------
// WMMA GEMM: C[M,N] (+)= A[M,K](fp16) * Bt[N,K](fp16, B transposed) (+ bias)
//   Wave computes a 32x64 tile: 2 A frags x 4 B frags -> 8 WMMAs per K-step.
//   Batched via blockIdx.z with element strides sA/sB/sC.
// Fragment layouts used here (contiguous k order, same for A and B):
//   A (16x32 f16): lane m=lane&15; elem e -> K = e + 16*(lane>>4)
//     => two contiguous half8 loads at row*K + k0 + 16*(lane>>4) and +8.
//   B (32x16 f16): lane col=lane&15; elem e -> K = e + 16*(lane>>4)
//     => one contiguous v16h load from Bt[col*K + k0 + 16*(lane>>4)].
//   C/D (16x16 f32): lane l, reg r -> row = r + 8*(l>>4), col = l&15.
// ---------------------------------------------------------------------------
__device__ __forceinline__ v16h load_a_frag(const _Float16* p) {
  half8 lo = *reinterpret_cast<const half8*>(p);
  half8 hi = *reinterpret_cast<const half8*>(p + 8);
  return __builtin_shufflevector(lo, hi, 0, 1, 2, 3, 4, 5, 6, 7,
                                 8, 9, 10, 11, 12, 13, 14, 15);
}

__global__ __launch_bounds__(128)
void wmma_gemm_kernel(const _Float16* __restrict__ A,
                      const _Float16* __restrict__ Bt,
                      float* __restrict__ Cmat,
                      const float* __restrict__ bias,
                      int M, int N, int K, int accum,
                      long long sA, long long sB, long long sC) {
  const int lane   = threadIdx.x & 31;
  const int wv     = blockIdx.x * 4 + (threadIdx.x >> 5);
  const int tilesN = N >> 6;
  const int tm     = wv / tilesN;  // 32-row tile index
  const int tn     = wv % tilesN;  // 64-col tile index
  if (tm * 32 >= M) return;  // wave-uniform exit: EXEC all-ones for WMMA

  const int hi  = lane >> 4;
  const int l15 = lane & 15;
  const long long zb = blockIdx.z;

  const _Float16* A0 = A + zb * sA + (long long)(tm * 32 + l15) * K + hi * 16;
  const _Float16* A1 = A0 + (long long)16 * K;
  const _Float16* Bb =
      Bt + zb * sB + (long long)(tn * 64 + l15) * K + hi * 16;

  v8f acc0[4] = {};
  v8f acc1[4] = {};
  for (int k0 = 0; k0 < K; k0 += 32) {
    const v16h af0 = load_a_frag(A0 + k0);
    const v16h af1 = load_a_frag(A1 + k0);
#pragma unroll
    for (int f = 0; f < 4; ++f) {
      const v16h bf =
          *reinterpret_cast<const v16h*>(Bb + (long long)f * 16 * K + k0);
      acc0[f] = __builtin_amdgcn_wmma_f32_16x16x32_f16(
          false, af0, false, bf, (short)0, acc0[f], false, false);
      acc1[f] = __builtin_amdgcn_wmma_f32_16x16x32_f16(
          false, af1, false, bf, (short)0, acc1[f], false, false);
    }
  }

  float* Cb = Cmat + zb * sC;
#pragma unroll
  for (int mi = 0; mi < 2; ++mi) {
#pragma unroll
    for (int f = 0; f < 4; ++f) {
      const int col = tn * 64 + f * 16 + l15;
      const float bv = bias ? bias[col] : 0.0f;
#pragma unroll
      for (int r = 0; r < 8; ++r) {
        const int row = tm * 32 + mi * 16 + r + hi * 8;
        const float v = (mi ? acc1[f][r] : acc0[f][r]) + bv;
        const long long idx = (long long)row * N + col;
        if (accum) Cb[idx] += v; else Cb[idx] = v;
      }
    }
  }
}

// ---------------------------------------------------------------------------
// ANNA: landmark centroids  cent[bh, m, d] = mean_{j<16} k[b,h,m*16+j,d]
// qkv layout: [b*1024+n, 1536], q at col h*64+d, k at +512, v at +1024
// ---------------------------------------------------------------------------
__global__ void anna_cent_kernel(const float* __restrict__ qkv,
                                 float* __restrict__ cent) {
  const int tid = blockIdx.x * blockDim.x + threadIdx.x;  // (bh*64+m)*64+d
  const int d  = tid & 63;
  const int m  = (tid >> 6) & 63;
  const int bh = tid >> 12;
  const int b = bh >> 3, h = bh & 7;
  const float* base =
      qkv + (long long)(b * 1024 + m * 16) * 1536 + 512 + h * 64 + d;
  float s = 0.f;
#pragma unroll
  for (int j = 0; j < 16; ++j) s += base[(long long)j * 1536];
  cent[tid] = s * (1.0f / 16.0f);
}

// ---------------------------------------------------------------------------
// ANNA attention: wave per query (b,h,n). Route to 64 landmarks, top-4 via
// iterative wave-argmax, softmax-attention over the 4*16 gathered keys.
// ---------------------------------------------------------------------------
__global__ __launch_bounds__(256)
void anna_attn_kernel(const float* __restrict__ qkv,
                      const float* __restrict__ cent,
                      _Float16* __restrict__ oah) {
  __shared__ float sq[8][64];
  __shared__ float sp[8][64];
  __shared__ int skn[8][64];
  const int lane = threadIdx.x & 31;
  const int w    = threadIdx.x >> 5;
  const int wid  = blockIdx.x * 8 + w;
  const int n    = wid & 1023;
  const int bh   = wid >> 10;
  const int b = bh >> 3, h = bh & 7;

  const long long qrow = (long long)(b * 1024 + n) * 1536;
  sq[w][lane]      = qkv[qrow + h * 64 + lane];
  sq[w][lane + 32] = qkv[qrow + h * 64 + lane + 32];
  __syncthreads();

  // landmark routing scores (2 landmarks per lane)
  float s0 = 0.f, s1 = 0.f;
  {
    const float* c0 = cent + ((long long)bh * 64 + lane) * 64;
    const float* c1 = c0 + 32 * 64;
    for (int d2 = 0; d2 < 64; ++d2) {
      const float qd = sq[w][d2];
      s0 += qd * c0[d2];
      s1 += qd * c1[d2];
    }
  }

  // top-4 (iterative wave argmax; ties -> smaller index, matches top_k)
  int kidx[4];
#pragma unroll
  for (int t = 0; t < 4; ++t) {
    float bv = s0; int bi = lane;
    if (s1 > bv) { bv = s1; bi = lane + 32; }
#pragma unroll
    for (int off = 16; off > 0; off >>= 1) {
      const float ov = __shfl_xor(bv, off, 32);
      const int   oi = __shfl_xor(bi, off, 32);
      if (ov > bv || (ov == bv && oi < bi)) { bv = ov; bi = oi; }
    }
    kidx[t] = bi;
    if (bi == lane) s0 = -1e30f;
    else if (bi == lane + 32) s1 = -1e30f;
  }

  // logits over 64 gathered keys (2 per lane); selection order irrelevant
  // after softmax + weighted sum (permutation invariant)
  const int nk0 = kidx[lane >> 4] * 16 + (lane & 15);
  const int nk1 = kidx[2 + (lane >> 4)] * 16 + (lane & 15);
  skn[w][lane] = nk0;
  skn[w][lane + 32] = nk1;
  float l0 = 0.f, l1 = 0.f;
  {
    const float* k0p = qkv + (long long)(b * 1024 + nk0) * 1536 + 512 + h * 64;
    const float* k1p = qkv + (long long)(b * 1024 + nk1) * 1536 + 512 + h * 64;
    for (int d2 = 0; d2 < 64; ++d2) {
      const float qd = sq[w][d2];
      l0 += qd * k0p[d2];
      l1 += qd * k1p[d2];
    }
  }
  l0 *= kScale; l1 *= kScale;

  float mx = fmaxf(l0, l1);
#pragma unroll
  for (int off = 16; off > 0; off >>= 1) mx = fmaxf(mx, __shfl_xor(mx, off, 32));
  const float e0 = __expf(l0 - mx), e1 = __expf(l1 - mx);
  float sm = e0 + e1;
#pragma unroll
  for (int off = 16; off > 0; off >>= 1) sm += __shfl_xor(sm, off, 32);
  const float inv = 1.0f / sm;
  sp[w][lane] = e0 * inv;
  sp[w][lane + 32] = e1 * inv;
  __syncthreads();

  // output: lane owns d=lane and d=lane+32
  float a0 = 0.f, a1 = 0.f;
  for (int j = 0; j < 64; ++j) {
    const float p = sp[w][j];
    const float* vp =
        qkv + (long long)(b * 1024 + skn[w][j]) * 1536 + 1024 + h * 64;
    a0 += p * vp[lane];
    a1 += p * vp[lane + 32];
  }
  const long long orow = (long long)(b * 1024 + n) * 512 + h * 64;
  oah[orow + lane]      = (_Float16)a0;
  oah[orow + lane + 32] = (_Float16)a1;
}

// ---------------------------------------------------------------------------
// AREA: 2x2 pooling of q/k/v. qa/ka in [bh][256 areas][64 d] (A and Bt of the
// scores GEMM); V emitted pre-transposed as vaT [bh][64 d][256 areas] so the
// PV GEMM also takes the contiguous Bt path.
// ---------------------------------------------------------------------------
__global__ void area_pool_kernel(const float* __restrict__ qkv,
                                 _Float16* __restrict__ qa,
                                 _Float16* __restrict__ ka,
                                 _Float16* __restrict__ vaT) {
  const int tid = blockIdx.x * blockDim.x + threadIdx.x;  // (bh*256+a)*64+d
  const int d  = tid & 63;
  const int a  = (tid >> 6) & 255;
  const int bh = tid >> 14;
  const int b = bh >> 3, h = bh & 7;
  const int r0 = (a >> 4) * 2, c0 = (a & 15) * 2;
  float fq = 0.f, fk = 0.f, fv = 0.f;
#pragma unroll
  for (int dy = 0; dy < 2; ++dy)
#pragma unroll
    for (int dx = 0; dx < 2; ++dx) {
      const int n = (r0 + dy) * 32 + c0 + dx;
      const float* base = qkv + (long long)(b * 1024 + n) * 1536 + h * 64 + d;
      fq += base[0];
      fk += base[512];
      fv += base[1024];
    }
  qa[tid] = (_Float16)(fq * 0.25f);
  ka[tid] = (_Float16)(fk * 0.25f);
  vaT[((long long)bh * 64 + d) * 256 + a] = (_Float16)(fv * 0.25f);
}

// Row softmax over 256 logits (wave per row); fp32 in, fp16 probs out.
__global__ __launch_bounds__(256)
void softmax256_kernel(const float* __restrict__ s, _Float16* __restrict__ p) {
  const int lane = threadIdx.x & 31;
  const int w    = threadIdx.x >> 5;
  const long long row = (long long)(blockIdx.x * 8 + w) * 256;
  float v[8];
  float mx = -1e30f;
#pragma unroll
  for (int t = 0; t < 8; ++t) {
    v[t] = s[row + t * 32 + lane] * kScale;
    mx = fmaxf(mx, v[t]);
  }
#pragma unroll
  for (int off = 16; off > 0; off >>= 1) mx = fmaxf(mx, __shfl_xor(mx, off, 32));
  float sm = 0.f;
#pragma unroll
  for (int t = 0; t < 8; ++t) { v[t] = __expf(v[t] - mx); sm += v[t]; }
#pragma unroll
  for (int off = 16; off > 0; off >>= 1) sm += __shfl_xor(sm, off, 32);
  const float inv = 1.0f / sm;
#pragma unroll
  for (int t = 0; t < 8; ++t)
    p[row + t * 32 + lane] = (_Float16)(v[t] * inv);
}

// Broadcast area outputs back to tokens with the reference's "faithful to
// torch" scrambled reshape: pos = h*65536 + n*64 + d -> (n'=pos>>9, c'=pos&511)
__global__ void area_scatter_kernel(const float* __restrict__ oa,
                                    _Float16* __restrict__ oah) {
  const int tid = blockIdx.x * blockDim.x + threadIdx.x;  // (bh*1024+n)*64+d
  const int d  = tid & 63;
  const int n  = (tid >> 6) & 1023;
  const int bh = tid >> 16;
  const int b = bh >> 3, h = bh & 7;
  const int row = n >> 5, col = n & 31;
  const int a = (row >> 1) * 16 + (col >> 1);
  const float val = oa[((long long)bh * 256 + a) * 64 + d];
  const int pos = h * 65536 + n * 64 + d;
  const int npr = pos >> 9, cpr = pos & 511;
  oah[((long long)b * 1024 + npr) * 512 + cpr] = (_Float16)val;
}

// ---------------------------------------------------------------------------
// GQA attention: wave per query, two-pass softmax with 1024 logits in LDS.
// Avoids materializing 32x1024x1024 score matrices (192MB ws / ~0.5GB HBM).
// ---------------------------------------------------------------------------
__global__ __launch_bounds__(256)
void gqa_attn_kernel(const float* __restrict__ gq, const float* __restrict__ gk,
                     const float* __restrict__ gv, _Float16* __restrict__ oah) {
  __shared__ float sq[8][64];
  __shared__ float sl[8][1024];
  const int lane = threadIdx.x & 31;
  const int w    = threadIdx.x >> 5;
  const int wid  = blockIdx.x * 8 + w;
  const int n    = wid & 1023;
  const int bh   = wid >> 10;
  const int b = bh >> 3, h = bh & 7, kvh = h >> 2;  // repeat_interleave(4)

  const long long qrow = (long long)(b * 1024 + n) * 512 + h * 64;
  sq[w][lane]      = gq[qrow + lane];
  sq[w][lane + 32] = gq[qrow + lane + 32];
  __syncthreads();

  float mx = -1e30f;
  for (int t = 0; t < 32; ++t) {
    const int j = t * 32 + lane;
    const float* kp = gk + (long long)(b * 1024 + j) * 128 + kvh * 64;
    float s = 0.f;
#pragma unroll
    for (int d2 = 0; d2 < 64; ++d2) s += sq[w][d2] * kp[d2];
    s *= kScale;
    sl[w][j] = s;
    mx = fmaxf(mx, s);
  }
#pragma unroll
  for (int off = 16; off > 0; off >>= 1) mx = fmaxf(mx, __shfl_xor(mx, off, 32));

  float sm = 0.f;
  for (int t = 0; t < 32; ++t) {
    const int j = t * 32 + lane;
    const float e = __expf(sl[w][j] - mx);
    sl[w][j] = e;
    sm += e;
  }
#pragma unroll
  for (int off = 16; off > 0; off >>= 1) sm += __shfl_xor(sm, off, 32);
  __syncthreads();

  float a0 = 0.f, a1 = 0.f;
  for (int j = 0; j < 1024; ++j) {
    const float p = sl[w][j];
    const float* vp = gv + (long long)(b * 1024 + j) * 128 + kvh * 64;
    a0 += p * vp[lane];
    a1 += p * vp[lane + 32];
  }
  const float inv = 1.0f / sm;
  const long long orow = (long long)(b * 1024 + n) * 512 + h * 64;
  oah[orow + lane]      = (_Float16)(a0 * inv);
  oah[orow + lane + 32] = (_Float16)(a1 * inv);
}

// ---------------------------------------------------------------------------
// Host launcher
// ---------------------------------------------------------------------------
extern "C" void kernel_launch(void* const* d_in, const int* in_sizes, int n_in,
                              void* d_out, int out_size, void* d_ws,
                              size_t ws_size, hipStream_t stream) {
  (void)in_sizes; (void)n_in; (void)out_size; (void)ws_size;
  const float* x       = (const float*)d_in[0];
  const float* aw_qkv  = (const float*)d_in[1];
  const float* aw_proj = (const float*)d_in[2];
  const float* ab_proj = (const float*)d_in[3];
  const float* rw_qkv  = (const float*)d_in[4];
  const float* rw_proj = (const float*)d_in[5];
  const float* rb_proj = (const float*)d_in[6];
  const float* gw_q    = (const float*)d_in[7];
  const float* gw_k    = (const float*)d_in[8];
  const float* gw_v    = (const float*)d_in[9];
  const float* gw_proj = (const float*)d_in[10];
  const float* gb_proj = (const float*)d_in[11];
  float* out = (float*)d_out;

  char* ws = (char*)d_ws;
  size_t off = 0;
  auto alloc = [&](size_t bytes) -> void* {
    void* p = ws + off;
    off = (off + bytes + 255) & ~(size_t)255;
    return p;
  };

  _Float16* xh  = (_Float16*)alloc(4096LL * 512 * 2);
  // all weights stored K-major ([N,K]) for contiguous WMMA B-fragment loads
  _Float16* wA  = (_Float16*)alloc(1536LL * 512 * 2);  // anna wqkv^T
  _Float16* wR  = (_Float16*)alloc(1536LL * 512 * 2);  // area wqkv^T
  _Float16* wGq = (_Float16*)alloc(512LL * 512 * 2);
  _Float16* wGk = (_Float16*)alloc(128LL * 512 * 2);
  _Float16* wGv = (_Float16*)alloc(128LL * 512 * 2);
  _Float16* wAp = (_Float16*)alloc(512LL * 512 * 2);
  _Float16* wRp = (_Float16*)alloc(512LL * 512 * 2);
  _Float16* wGp = (_Float16*)alloc(512LL * 512 * 2);
  float* qkv    = (float*)alloc(4096LL * 1536 * 4);  // shared anna/area qkv
  float* gq     = (float*)alloc(4096LL * 512 * 4);
  float* gk     = (float*)alloc(4096LL * 128 * 4);
  float* gv     = (float*)alloc(4096LL * 128 * 4);
  float* cent   = (float*)alloc(2048LL * 64 * 4);
  _Float16* qa  = (_Float16*)alloc(32LL * 256 * 64 * 2);
  _Float16* ka  = (_Float16*)alloc(32LL * 256 * 64 * 2);
  _Float16* vaT = (_Float16*)alloc(32LL * 64 * 256 * 2);
  float* scores = (float*)alloc(32LL * 256 * 256 * 4);
  _Float16* prb = (_Float16*)alloc(32LL * 256 * 256 * 2);
  float* oaarea = (float*)alloc(32LL * 256 * 64 * 4);
  _Float16* oah = (_Float16*)alloc(4096LL * 512 * 2);  // shared head output
  float* acc    = (float*)alloc(4096LL * 512 * 4);

  auto cvt = [&](const float* s, _Float16* d, long long n) {
    int blocks = (int)((n + 255) / 256);
    if (blocks > 8192) blocks = 8192;
    cvt_f16_kernel<<<dim3(blocks), dim3(256), 0, stream>>>(s, d, n);
  };
  auto cvtT = [&](const float* s, _Float16* d, int K, int N) {
    const long long total = (long long)K * N;
    const int blocks = (int)((total + 255) / 256);
    cvt_t_f16_kernel<<<dim3(blocks), dim3(256), 0, stream>>>(s, d, K, N);
  };
  auto gemm = [&](const _Float16* A, const _Float16* Bt, float* Cm,
                  const float* bias, int M, int N, int K, int accum,
                  long long sA, long long sB, long long sC, int batch) {
    const int tiles = (M / 32) * (N / 64);
    const int blocks = (tiles + 3) / 4;
    wmma_gemm_kernel<<<dim3(blocks, 1, batch), dim3(128), 0, stream>>>(
        A, Bt, Cm, bias, M, N, K, accum, sA, sB, sC);
  };

  // fp16 activations + K-major fp16 weights
  cvt(x, xh, 4096LL * 512);
  cvtT(aw_qkv, wA, 512, 1536);
  cvtT(rw_qkv, wR, 512, 1536);
  cvtT(gw_q, wGq, 512, 512);
  cvtT(gw_k, wGk, 512, 128);
  cvtT(gw_v, wGv, 512, 128);
  cvtT(aw_proj, wAp, 512, 512);
  cvtT(rw_proj, wRp, 512, 512);
  cvtT(gw_proj, wGp, 512, 512);

  zero_kernel<<<dim3(8192), dim3(256), 0, stream>>>(acc, 4096LL * 512);

  // ---- ANNA branch ----
  gemm(xh, wA, qkv, nullptr, 4096, 1536, 512, 0, 0, 0, 0, 1);
  anna_cent_kernel<<<dim3(512), dim3(256), 0, stream>>>(qkv, cent);
  anna_attn_kernel<<<dim3(4096), dim3(256), 0, stream>>>(qkv, cent, oah);
  gemm(oah, wAp, acc, ab_proj, 4096, 512, 512, 1, 0, 0, 0, 1);

  // ---- AREA branch ----
  gemm(xh, wR, qkv, nullptr, 4096, 1536, 512, 0, 0, 0, 0, 1);
  area_pool_kernel<<<dim3(2048), dim3(256), 0, stream>>>(qkv, qa, ka, vaT);
  gemm(qa, ka, scores, nullptr, 256, 256, 64, 0, 256LL * 64, 256LL * 64,
       256LL * 256, 32);
  softmax256_kernel<<<dim3(1024), dim3(256), 0, stream>>>(scores, prb);
  gemm(prb, vaT, oaarea, nullptr, 256, 64, 256, 0, 256LL * 256, 64LL * 256,
       256LL * 64, 32);
  area_scatter_kernel<<<dim3(8192), dim3(256), 0, stream>>>(oaarea, oah);
  gemm(oah, wRp, acc, rb_proj, 4096, 512, 512, 1, 0, 0, 0, 1);

  // ---- GQA branch ----
  gemm(xh, wGq, gq, nullptr, 4096, 512, 512, 0, 0, 0, 0, 1);
  gemm(xh, wGk, gk, nullptr, 4096, 128, 512, 0, 0, 0, 0, 1);
  gemm(xh, wGv, gv, nullptr, 4096, 128, 512, 0, 0, 0, 0, 1);
  gqa_attn_kernel<<<dim3(4096), dim3(256), 0, stream>>>(gq, gk, gv, oah);
  gemm(oah, wGp, acc, gb_proj, 4096, 512, 512, 1, 0, 0, 0, 1);

  // ---- combine ----
  finalize_kernel<<<dim3(8192), dim3(256), 0, stream>>>(acc, out,
                                                        4096LL * 512);
}
